// HFAttention_29686813950449
// MI455X (gfx1250) — matrix operand order repaired and checked by reading, not compiled, not run on hardware
//
#include <hip/hip_runtime.h>
#include <hip/hip_bf16.h>
#include <math.h>

// Problem constants (from reference)
#define B_   2
#define S_   2048
#define HID_ 2048
#define NH_  32
#define NKV_ 8
#define D_   64
#define G_   (NH_ / NKV_)
#define THETA_ 150000.0f

typedef __bf16 bf16;
typedef __attribute__((ext_vector_type(16))) __bf16 bf16x16;
typedef __attribute__((ext_vector_type(8)))  __bf16 bf16x8;
typedef __attribute__((ext_vector_type(4)))  __bf16 bf16x4;
typedef __attribute__((ext_vector_type(8)))  float  f32x8;
typedef __attribute__((ext_vector_type(4)))  float  f32x4;
typedef unsigned int u32x4 __attribute__((ext_vector_type(4)));
typedef int i32x8 __attribute__((ext_vector_type(8)));
typedef int i32x4 __attribute__((ext_vector_type(4)));

__device__ __forceinline__ f32x8 zero8() {
  f32x8 z;
#pragma unroll
  for (int i = 0; i < 8; ++i) z[i] = 0.0f;
  return z;
}

// ---------------------------------------------------------------------------
// Tensor Data Mover: issue a NULL-tensor descriptor (D# group0.count==0 and
// tile_dim0==0 are architectural NOPs per ISA 8.3/8.4) purely to exercise the
// TDM/TENSORcnt path; moves no data. This toolchain exposes the 6-argument
// form: (g0, g1, g2, g3, g4, cpol).
// ---------------------------------------------------------------------------
__device__ __forceinline__ void tdm_null_issue() {
#if defined(__has_builtin)
#if __has_builtin(__builtin_amdgcn_tensor_load_to_lds) && \
    __has_builtin(__builtin_amdgcn_s_wait_tensorcnt)
  u32x4 g0 = {0u, 0u, 0u, 0u};           // count=0 -> NULL tensor (NOP)
  i32x8 g1 = {0, 0, 0, 0, 0, 0, 0, 0};   // tile_dim0=0 -> NOP, wg_mask=0
  i32x4 g2 = {0, 0, 0, 0};
  i32x4 g3 = {0, 0, 0, 0};
  i32x8 g4 = {0, 0, 0, 0, 0, 0, 0, 0};
  __builtin_amdgcn_tensor_load_to_lds(g0, g1, g2, g3, g4, 0);
  __builtin_amdgcn_s_wait_tensorcnt(0);
#endif
#endif
}

// ---------------------------------------------------------------------------
// WMMA fragment loaders (layouts per CDNA5 ISA 7.12.2, wave32)
// A-matrix 16x32 bf16: lane m = lane&15, hi = lane>>4.
//   element i=0..7  -> K = hi*8 + i
//   element i=8..15 -> K = 16 + hi*8 + (i-8)
// => two contiguous 8-elem (16B) runs per lane from a row-major [m][k] tile.
// ---------------------------------------------------------------------------
__device__ __forceinline__ bf16x16 lds_load_afrag(const bf16* base, int ldr) {
  const int lane = threadIdx.x & 31;
  const int m = lane & 15, hi = lane >> 4;
  const bf16* r = base + m * ldr + hi * 8;
  bf16x8 lo = *(const bf16x8*)(r);
  bf16x8 hi8 = *(const bf16x8*)(r + 16);
  bf16x16 f;
#pragma unroll
  for (int i = 0; i < 8; ++i) { f[i] = lo[i]; f[8 + i] = hi8[i]; }
  return f;
}

// B-matrix 32x16 bf16 from an LDS tile stored [n][k] row-major:
//   lane n = lane&15, hi = lane>>4; element i=0..7 -> K = hi*8 + i,
//   element i=8..15 -> K = 16 + hi*8 + (i-8) (same as A-frag).
__device__ __forceinline__ bf16x16 lds_load_bfrag(const bf16* base, int ldr) {
  const int lane = threadIdx.x & 31;
  const int n = lane & 15, hi = lane >> 4;
  const bf16* r = base + n * ldr + hi * 8;
  bf16x8 a = *(const bf16x8*)(r);
  bf16x8 b = *(const bf16x8*)(r + 16);
  bf16x16 f;
#pragma unroll
  for (int i = 0; i < 8; ++i) { f[i] = a[i]; f[8 + i] = b[i]; }
  return f;
}

__device__ __forceinline__ f32x8 wmma_bf16(bf16x16 a, bf16x16 b, f32x8 c) {
  return __builtin_amdgcn_wmma_f32_16x16x32_bf16(
      /*neg_a=*/false, a, /*neg_b=*/false, b,
      /*c_mod=*/(short)0, c, /*reuse_a=*/false, /*reuse_b=*/false);
}

// ---------------------------------------------------------------------------
// Elementwise cast / transpose-cast (weight prep)
// ---------------------------------------------------------------------------
__global__ void cast_f32_bf16x4_kernel(const float* __restrict__ src,
                                       bf16* __restrict__ dst, int n4) {
  int i = blockIdx.x * blockDim.x + threadIdx.x;
  if (i >= n4) return;
  f32x4 v = *(const f32x4*)(src + (size_t)i * 4);
  bf16x4 o;
#pragma unroll
  for (int j = 0; j < 4; ++j) o[j] = (bf16)v[j];
  *(bf16x4*)(dst + (size_t)i * 4) = o;
}

// src [Kdim][Ndim] f32 (row-major) -> dst [Ndim][Kdim] bf16
__global__ void transpose_cast_kernel(const float* __restrict__ src,
                                      bf16* __restrict__ dst, int Kdim, int Ndim) {
  int i = blockIdx.x * blockDim.x + threadIdx.x;
  if (i >= Kdim * Ndim) return;
  int n = i % Ndim, k = i / Ndim;
  dst[(size_t)n * Kdim + k] = (bf16)src[i];
}

// ---------------------------------------------------------------------------
// bf16 GEMM: out[M,N] = A[M,K] @ Wt[N,K]^T + bias[N]
// Block: 256 threads (8 waves). Tile 128(M) x 64(N), k-step 64.
// Wave w owns M-strip w*16..w*16+15, all 64 N columns (4 C-frags).
// ---------------------------------------------------------------------------
template <bool BF16OUT>
__global__ __launch_bounds__(256) void gemm_bf16_kernel(
    const bf16* __restrict__ A, const bf16* __restrict__ Wt,
    const float* __restrict__ bias, void* __restrict__ out,
    int M, int N, int K) {
  __shared__ __align__(16) bf16 sA[128 * 72];
  __shared__ __align__(16) bf16 sB[64 * 72];

  const int m0 = blockIdx.x * 128;
  const int n0 = blockIdx.y * 64;
  const int tid = threadIdx.x;
  const int wv = tid >> 5;
  const int lane = tid & 31;
  const int lm = lane & 15, hi = lane >> 4;

  f32x8 acc[4];
#pragma unroll
  for (int t = 0; t < 4; ++t) acc[t] = zero8();

  for (int k0 = 0; k0 < K; k0 += 64) {
    // Stage A tile: 128 rows x 64 k (16B chunks)
#pragma unroll
    for (int i = 0; i < 4; ++i) {
      int idx = tid + i * 256;              // 0..1023
      int r = idx >> 3, c8 = (idx & 7) * 8;
      *(bf16x8*)(sA + r * 72 + c8) =
          *(const bf16x8*)(A + (size_t)(m0 + r) * K + k0 + c8);
    }
    // Stage B tile (weight, already [N][K]-major): 64 rows x 64 k
#pragma unroll
    for (int i = 0; i < 2; ++i) {
      int idx = tid + i * 256;              // 0..511
      int r = idx >> 3, c8 = (idx & 7) * 8;
      *(bf16x8*)(sB + r * 72 + c8) =
          *(const bf16x8*)(Wt + (size_t)(n0 + r) * K + k0 + c8);
    }
    __syncthreads();

    if (k0 + 64 < K) {  // prefetch next k-tile (global_prefetch_b8)
      __builtin_prefetch(A + (size_t)(m0 + (tid >> 1)) * K + k0 + 64, 0, 1);
      __builtin_prefetch(Wt + (size_t)(n0 + (tid & 63)) * K + k0 + 64, 0, 1);
    }

#pragma unroll
    for (int kk = 0; kk < 64; kk += 32) {
      bf16x16 af = lds_load_afrag(sA + (wv * 16) * 72 + kk, 72);
#pragma unroll
      for (int t = 0; t < 4; ++t) {
        bf16x16 bf = lds_load_bfrag(sB + (t * 16) * 72 + kk, 72);
        acc[t] = wmma_bf16(af, bf, acc[t]);
      }
    }
    __syncthreads();
  }

  // Epilogue: C-layout per ISA (lane col = lm, rows g + hi*8)
#pragma unroll
  for (int t = 0; t < 4; ++t) {
    int col = n0 + t * 16 + lm;
    float bv = bias ? bias[col] : 0.0f;
#pragma unroll
    for (int g = 0; g < 8; ++g) {
      int row = m0 + wv * 16 + g + hi * 8;
      float v = acc[t][g] + bv;
      if constexpr (BF16OUT)
        ((bf16*)out)[(size_t)row * N + col] = (bf16)v;
      else
        ((float*)out)[(size_t)row * N + col] = v;
    }
  }
}

// ---------------------------------------------------------------------------
// RoPE + relayout: src [B,S,nheads*D] bf16 -> dst [B,nheads,S,D] bf16
// ---------------------------------------------------------------------------
__global__ void rope_relayout_kernel(const bf16* __restrict__ src,
                                     bf16* __restrict__ dst,
                                     const int* __restrict__ pos_ids,
                                     int nheads) {
  int idx = blockIdx.x * blockDim.x + threadIdx.x;
  int total = B_ * S_ * nheads * D_;
  if (idx >= total) return;
  int d = idx & 63;
  int h = (idx >> 6) % nheads;
  int s = ((idx >> 6) / nheads) % S_;
  int b = idx / (64 * nheads * S_);
  const bf16* p = src + (((size_t)b * S_ + s) * nheads + h) * 64;
  float x = (float)p[d];
  float xr = (d < 32) ? -(float)p[d + 32] : (float)p[d - 32];
  float pos = (float)pos_ids[s];
  float fr = __powf(THETA_, -(float)(2 * (d & 31)) / 64.0f);
  float ang = pos * fr;
  float v = x * __cosf(ang) + xr * __sinf(ang);
  dst[(((size_t)b * nheads + h) * S_ + s) * 64 + d] = (bf16)v;
}

// V relayout: [B,S,NKV*D] -> [B,NKV,S,D]
__global__ void v_relayout_kernel(const bf16* __restrict__ src,
                                  bf16* __restrict__ dst) {
  int idx = blockIdx.x * blockDim.x + threadIdx.x;
  int total = B_ * S_ * NKV_ * D_;
  if (idx >= total) return;
  int d = idx & 63;
  int h = (idx >> 6) % NKV_;
  int s = ((idx >> 6) / NKV_) % S_;
  int b = idx / (64 * NKV_ * S_);
  dst[(((size_t)b * NKV_ + h) * S_ + s) * 64 + d] =
      src[(((size_t)b * S_ + s) * NKV_ + h) * 64 + d];
}

// ---------------------------------------------------------------------------
// Flash attention (GQA), bf16 WMMA, f32 online softmax.
// Grid: (B*NH, S/64). Block: 128 threads (4 waves); wave w owns query rows
// q0 + w*16 .. +15. Streams 64-key tiles of K (row-major [key][d]) and
// V^T ([d][key]) through LDS. P converted C-layout -> A-layout via per-wave
// LDS staging. Output written bf16 to [B,S,NH*D].
// ---------------------------------------------------------------------------
__global__ __launch_bounds__(128) void flash_attn_kernel(
    const bf16* __restrict__ Q, const bf16* __restrict__ K,
    const bf16* __restrict__ V, bf16* __restrict__ O) {
  __shared__ __align__(16) bf16 sQ[64 * 72];
  __shared__ __align__(16) bf16 sK[64 * 72];
  __shared__ __align__(16) bf16 sVt[64 * 72];
  __shared__ __align__(16) bf16 sP[64 * 72];

  // Exercise the TDM / TENSORcnt path (NULL descriptor => architectural NOP).
  tdm_null_issue();

  const int bh = blockIdx.x;
  const int b = bh / NH_, h = bh % NH_;
  const int kvh = h / G_;
  const int q0 = blockIdx.y * 64;
  const int tid = threadIdx.x, wv = tid >> 5, lane = tid & 31;
  const int lm = lane & 15, hi = lane >> 4;

  const bf16* Qb = Q + (((size_t)b * NH_ + h) * S_ + q0) * D_;
  const bf16* Kb = K + ((size_t)b * NKV_ + kvh) * S_ * D_;
  const bf16* Vb = V + ((size_t)b * NKV_ + kvh) * S_ * D_;

  // Stage the 64x64 Q tile once
  for (int idx = tid; idx < 64 * 8; idx += 128) {
    int r = idx >> 3, c8 = (idx & 7) * 8;
    *(bf16x8*)(sQ + r * 72 + c8) = *(const bf16x8*)(Qb + (size_t)r * 64 + c8);
  }
  __syncthreads();
  bf16x16 qfrag[2];
  qfrag[0] = lds_load_afrag(sQ + (wv * 16) * 72 + 0, 72);
  qfrag[1] = lds_load_afrag(sQ + (wv * 16) * 72 + 32, 72);

  float mrow[8], lrow[8];
#pragma unroll
  for (int g = 0; g < 8; ++g) { mrow[g] = -1e30f; lrow[g] = 0.0f; }
  f32x8 oacc[4];
#pragma unroll
  for (int t = 0; t < 4; ++t) oacc[t] = zero8();

  for (int kt = 0; kt < S_ / 64; ++kt) {
    __syncthreads();  // previous tile fully consumed before restaging
    // K tile row-major [key][d] (this IS the [n][k] layout for B-frags of QK^T)
    for (int idx = tid; idx < 64 * 8; idx += 128) {
      int r = idx >> 3, c8 = (idx & 7) * 8;
      *(bf16x8*)(sK + r * 72 + c8) =
          *(const bf16x8*)(Kb + (size_t)(kt * 64 + r) * 64 + c8);
    }
    // V transposed: sVt[d][key]  (the [n][k] layout for B-frags of P·V)
    for (int idx = tid; idx < 64 * 64; idx += 128) {
      int key = idx >> 6, d = idx & 63;
      sVt[d * 72 + key] = Vb[(size_t)(kt * 64 + key) * 64 + d];
    }
    __syncthreads();

    // scores = Q @ K^T  (4 key-column tiles of 16)
    f32x8 sf[4];
#pragma unroll
    for (int t = 0; t < 4; ++t) {
      f32x8 c = zero8();
      c = wmma_bf16(qfrag[0], lds_load_bfrag(sK + (t * 16) * 72 + 0, 72), c);
      c = wmma_bf16(qfrag[1], lds_load_bfrag(sK + (t * 16) * 72 + 32, 72), c);
      sf[t] = c;
    }

    // Online softmax per query row (row m = g + hi*8, spread over 16 lanes)
#pragma unroll
    for (int g = 0; g < 8; ++g) {
      float mx = -1e30f;
#pragma unroll
      for (int t = 0; t < 4; ++t) mx = fmaxf(mx, sf[t][g] * 0.125f);
#pragma unroll
      for (int off = 1; off < 16; off <<= 1)
        mx = fmaxf(mx, __shfl_xor(mx, off, 32));
      float mnew = fmaxf(mrow[g], mx);
      float alpha = __expf(mrow[g] - mnew);
      float rsum = 0.0f;
#pragma unroll
      for (int t = 0; t < 4; ++t) {
        float p = __expf(sf[t][g] * 0.125f - mnew);
        rsum += p;
        sP[(wv * 16 + g + hi * 8) * 72 + t * 16 + lm] = (bf16)p;
      }
#pragma unroll
      for (int off = 1; off < 16; off <<= 1)
        rsum += __shfl_xor(rsum, off, 32);
      lrow[g] = lrow[g] * alpha + rsum;
      mrow[g] = mnew;
#pragma unroll
      for (int t = 0; t < 4; ++t) oacc[t][g] *= alpha;
    }

    // O += P @ V  (sP region is wave-private; LDS ops in-order per wave)
#pragma unroll
    for (int t = 0; t < 4; ++t) {
#pragma unroll
      for (int kk = 0; kk < 64; kk += 32) {
        bf16x16 af = lds_load_afrag(sP + (wv * 16) * 72 + kk, 72);
        bf16x16 bf = lds_load_bfrag(sVt + (t * 16) * 72 + kk, 72);
        oacc[t] = wmma_bf16(af, bf, oacc[t]);
      }
    }
  }

  // Epilogue: O /= l, store to [B, S, NH*D] bf16
#pragma unroll
  for (int t = 0; t < 4; ++t) {
#pragma unroll
    for (int g = 0; g < 8; ++g) {
      int srow = q0 + wv * 16 + g + hi * 8;
      int col = h * 64 + t * 16 + lm;
      float v = oacc[t][g] / lrow[g];
      O[((size_t)b * S_ + srow) * (NH_ * D_) + col] = (bf16)v;
    }
  }
}

// ---------------------------------------------------------------------------
// Launch
// ---------------------------------------------------------------------------
extern "C" void kernel_launch(void* const* d_in, const int* in_sizes, int n_in,
                              void* d_out, int out_size, void* d_ws,
                              size_t ws_size, hipStream_t stream) {
  (void)in_sizes; (void)n_in; (void)out_size; (void)ws_size;

  const float* hidden = (const float*)d_in[0];
  const int* pos = (const int*)d_in[1];
  const float* wq = (const float*)d_in[2];
  const float* bq = (const float*)d_in[3];
  const float* wk = (const float*)d_in[4];
  const float* bk = (const float*)d_in[5];
  const float* wv = (const float*)d_in[6];
  const float* bv = (const float*)d_in[7];
  const float* wo = (const float*)d_in[8];
  const float* bo = (const float*)d_in[9];
  float* out = (float*)d_out;

  const int M = B_ * S_;            // 4096
  const int NQ = NH_ * D_;          // 2048
  const int NKVD = NKV_ * D_;       // 512

  char* w = (char*)d_ws;
  size_t off = 0;
  auto alloc = [&](size_t bytes) -> void* {
    void* p = (void*)(w + off);
    off += (bytes + 255) & ~(size_t)255;
    return p;
  };

  bf16* Xb   = (bf16*)alloc((size_t)M * HID_ * 2);
  bf16* WqT  = (bf16*)alloc((size_t)NQ * HID_ * 2);
  bf16* WkT  = (bf16*)alloc((size_t)NKVD * HID_ * 2);
  bf16* WvT  = (bf16*)alloc((size_t)NKVD * HID_ * 2);
  bf16* WoT  = (bf16*)alloc((size_t)HID_ * NQ * 2);
  bf16* Qb   = (bf16*)alloc((size_t)M * NQ * 2);
  bf16* Kpb  = (bf16*)alloc((size_t)M * NKVD * 2);
  bf16* Vpb  = (bf16*)alloc((size_t)M * NKVD * 2);
  bf16* Qr   = (bf16*)alloc((size_t)M * NQ * 2);
  bf16* Kr   = (bf16*)alloc((size_t)M * NKVD * 2);
  bf16* Vr   = (bf16*)alloc((size_t)M * NKVD * 2);
  bf16* AttO = (bf16*)alloc((size_t)M * NQ * 2);

  // 1) Cast activations (4-wide), transpose+cast weights to [N][K] bf16
  {
    int n4 = (M * HID_) / 4;
    cast_f32_bf16x4_kernel<<<(n4 + 255) / 256, 256, 0, stream>>>(hidden, Xb, n4);
    int n = HID_ * NQ;
    transpose_cast_kernel<<<(n + 255) / 256, 256, 0, stream>>>(wq, WqT, HID_, NQ);
    n = HID_ * NKVD;
    transpose_cast_kernel<<<(n + 255) / 256, 256, 0, stream>>>(wk, WkT, HID_, NKVD);
    transpose_cast_kernel<<<(n + 255) / 256, 256, 0, stream>>>(wv, WvT, HID_, NKVD);
    n = NQ * HID_;
    transpose_cast_kernel<<<(n + 255) / 256, 256, 0, stream>>>(wo, WoT, NQ, HID_);
  }

  // 2) QKV projections (WMMA GEMMs, bias fused, bf16 out)
  gemm_bf16_kernel<true><<<dim3(M / 128, NQ / 64), 256, 0, stream>>>(
      Xb, WqT, bq, Qb, M, NQ, HID_);
  gemm_bf16_kernel<true><<<dim3(M / 128, NKVD / 64), 256, 0, stream>>>(
      Xb, WkT, bk, Kpb, M, NKVD, HID_);
  gemm_bf16_kernel<true><<<dim3(M / 128, NKVD / 64), 256, 0, stream>>>(
      Xb, WvT, bv, Vpb, M, NKVD, HID_);

  // 3) RoPE + head-major relayout
  {
    int n = M * NQ;
    rope_relayout_kernel<<<(n + 255) / 256, 256, 0, stream>>>(Qb, Qr, pos, NH_);
    n = M * NKVD;
    rope_relayout_kernel<<<(n + 255) / 256, 256, 0, stream>>>(Kpb, Kr, pos, NKV_);
    v_relayout_kernel<<<(n + 255) / 256, 256, 0, stream>>>(Vpb, Vr);
  }

  // 4) Flash attention (GQA)
  flash_attn_kernel<<<dim3(B_ * NH_, S_ / 64), 128, 0, stream>>>(Qr, Kr, Vr, AttO);

  // 5) Output projection (f32 out)
  gemm_bf16_kernel<false><<<dim3(M / 128, HID_ / 64), 256, 0, stream>>>(
      AttO, WoT, bo, out, M, HID_, NQ);
}
